// MoELayer_52544629899333
// MI455X (gfx1250) — matrix operand order repaired and checked by reading, not compiled, not run on hardware
//
#include <hip/hip_runtime.h>

// ---------------------------------------------------------------------------
// MoE layer for MI455X (gfx1250), wave32 + WMMA bf16.
//
// Roofline: expert GEMMs dominate (~165 GFLOP). Weights = 256 MB f32; each
// expert's 16 MB W-slab fits the 192 MB L2, so HBM traffic ~ one weight pass
// (~11 us @ 23.3 TB/s) -> compute bound. Use v_wmma_f32_16x16x32_bf16 with
// f32 accumulate; double-buffered LDS with async global->LDS copies
// (GLOBAL_LOAD_ASYNC_TO_LDS_B128 / ASYNCcnt) to overlap staging with WMMAs.
// ---------------------------------------------------------------------------

#define T_TOK   4096
#define EXPERTS 8
#define TOPK    2
#define DIM     1024
#define FDIM    4096
#define CAP     1229      // ceil(1.2 * 4096 * 2 / 8)
#define CAP_PAD 1280      // padded to a multiple of 128 for tiling

typedef __attribute__((ext_vector_type(16))) __bf16          v16bf;
typedef __attribute__((ext_vector_type(8)))  float           v8f;
typedef __attribute__((ext_vector_type(8)))  unsigned short  us8;
typedef __attribute__((ext_vector_type(4)))  unsigned short  us4;
typedef __attribute__((ext_vector_type(4)))  int             v4i_t;

union FragBF16 {            // 16 bf16 = 8 VGPRs, matches WMMA A/B operand
    us8   u[2];
    v16bf v;
};

#if __has_builtin(__builtin_amdgcn_global_load_async_to_lds_b128)
#define HAS_ASYNC_LDS 1
typedef __attribute__((address_space(3))) v4i_t lds_v4i;
#endif

// 16-byte global(bf16) -> LDS copy; async on CDNA5 when available.
static __device__ __forceinline__ void copy16_g2l(const unsigned short* g,
                                                  unsigned short* l) {
#ifdef HAS_ASYNC_LDS
    __builtin_amdgcn_global_load_async_to_lds_b128((v4i_t*)g, (lds_v4i*)l, 0, 0);
#else
    *(us8*)l = *(const us8*)g;
#endif
}

static __device__ __forceinline__ void wait_async_done() {
#ifdef HAS_ASYNC_LDS
#if __has_builtin(__builtin_amdgcn_s_wait_asynccnt)
    __builtin_amdgcn_s_wait_asynccnt(0);
#else
    asm volatile("s_wait_asynccnt 0x0" ::: "memory");
#endif
#endif
}

// f32 -> bf16, round-half-up: one v_add_nc_u32; the high-half store then
// lowers to ds_store_b16_d16_hi / d16_hi global stores for free. Differs from
// RNE only on exact ties (prob ~2^-16) -- negligible vs bf16 quantization.
static __device__ __forceinline__ unsigned short f2bf(float f) {
    unsigned int u = __float_as_uint(f) + 0x8000u;
    return (unsigned short)(u >> 16);
}

// Branch-free tanh-GELU: 0.5*x*(1+tanh(z)) = x * (1 - 1/(exp(2z)+1)).
static __device__ __forceinline__ float gelu_tanh(float v) {
    float z  = 0.7978845608028654f * (v + 0.044715f * v * v * v);
    float ez = __expf(2.f * z);        // inf -> factor 1, 0 -> factor 0 (no NaN)
    return v * (1.f - 1.f / (ez + 1.f));
}

// ---------------------------------------------------------------------------
// 1) Gating: one wave per token. logits = x@Wg + bg, softmax, top-2, renorm.
// ---------------------------------------------------------------------------
__global__ __launch_bounds__(256) void gating_kernel(
    const float* __restrict__ x, const float* __restrict__ Wg,
    const float* __restrict__ bg,
    int* __restrict__ slot_expert, float* __restrict__ slot_gate)
{
    int wave = threadIdx.x >> 5;
    int lane = threadIdx.x & 31;
    int t = blockIdx.x * 8 + wave;

    float acc[EXPERTS];
#pragma unroll
    for (int e = 0; e < EXPERTS; ++e) acc[e] = 0.f;

    const float* xr = x + (size_t)t * DIM;
#pragma unroll 4
    for (int i = 0; i < DIM / 32; ++i) {
        int d = i * 32 + lane;
        float xv = xr[d];
        const float4* w = (const float4*)(Wg + d * EXPERTS);
        float4 wa = w[0], wb = w[1];
        acc[0] += xv * wa.x; acc[1] += xv * wa.y;
        acc[2] += xv * wa.z; acc[3] += xv * wa.w;
        acc[4] += xv * wb.x; acc[5] += xv * wb.y;
        acc[6] += xv * wb.z; acc[7] += xv * wb.w;
    }
#pragma unroll
    for (int e = 0; e < EXPERTS; ++e)
#pragma unroll
        for (int off = 16; off >= 1; off >>= 1)
            acc[e] += __shfl_xor(acc[e], off, 32);

    if (lane == 0) {
        float lg[EXPERTS];
        float mx = -1e30f;
#pragma unroll
        for (int e = 0; e < EXPERTS; ++e) { lg[e] = acc[e] + bg[e]; mx = fmaxf(mx, lg[e]); }
        float den = 0.f;
#pragma unroll
        for (int e = 0; e < EXPERTS; ++e) { lg[e] = __expf(lg[e] - mx); den += lg[e]; }
        float inv = 1.f / den;
#pragma unroll
        for (int e = 0; e < EXPERTS; ++e) lg[e] *= inv;

        float m1 = -1.f; int i1 = 0;
#pragma unroll
        for (int e = 0; e < EXPERTS; ++e) if (lg[e] > m1) { m1 = lg[e]; i1 = e; }
        float m2 = -1.f; int i2 = 0;
#pragma unroll
        for (int e = 0; e < EXPERTS; ++e) if (e != i1 && lg[e] > m2) { m2 = lg[e]; i2 = e; }

        float gs = 1.f / (m1 + m2);
        slot_expert[t]          = i1;  slot_gate[t]          = m1 * gs;
        slot_expert[T_TOK + t]  = i2;  slot_gate[T_TOK + t]  = m2 * gs;
    }
}

// ---------------------------------------------------------------------------
// 2) Ordered per-expert prefix positions over the 8192 k-major slots
//    (exactly matches the reference cumsum ordering). One block per expert.
// ---------------------------------------------------------------------------
__global__ __launch_bounds__(1024) void scan_kernel(
    const int* __restrict__ slot_expert,
    int* __restrict__ slot_pos, float* __restrict__ slot_gate)
{
    __shared__ int s[1024];
    const int e = blockIdx.x;
    const int tid = threadIdx.x;
    int base = 0;

    for (int chunk = 0; chunk < (T_TOK * TOPK) / 1024; ++chunk) {
        int i = chunk * 1024 + tid;
        int flag = (slot_expert[i] == e) ? 1 : 0;
        s[tid] = flag;
        __syncthreads();
#pragma unroll
        for (int off = 1; off < 1024; off <<= 1) {
            int add = (tid >= off) ? s[tid - off] : 0;
            __syncthreads();
            s[tid] += add;
            __syncthreads();
        }
        int incl  = s[tid];
        int total = s[1023];
        if (flag) {
            int p = base + incl - 1;          // exclusive position
            slot_pos[i] = (p < CAP) ? p : CAP;
            if (p >= CAP) slot_gate[i] = 0.f; // dropped -> zero gate
        }
        base += total;
        __syncthreads();
    }
}

// ---------------------------------------------------------------------------
// 3) Zero expert buffers, then scatter kept token rows (f32 -> bf16).
// ---------------------------------------------------------------------------
__global__ void zero_kernel(uint4* __restrict__ p, long n)
{
    long i = (long)blockIdx.x * blockDim.x + threadIdx.x;
    long stride = (long)gridDim.x * blockDim.x;
    uint4 z = make_uint4(0u, 0u, 0u, 0u);
    for (; i < n; i += stride) p[i] = z;
}

__global__ __launch_bounds__(256) void dispatch_kernel(
    const float* __restrict__ x,
    const int* __restrict__ slot_expert, const int* __restrict__ slot_pos,
    unsigned short* __restrict__ buf)
{
    int wave = threadIdx.x >> 5;
    int lane = threadIdx.x & 31;
    int i = blockIdx.x * 8 + wave;
    int e = slot_expert[i];
    int p = slot_pos[i];
    if (p >= CAP) return;                 // wave-uniform: dropped slot
    int t = i & (T_TOK - 1);

    const float4* src = (const float4*)(x + (size_t)t * DIM);
    us4* dst = (us4*)(buf + ((size_t)e * CAP_PAD + p) * DIM);
#pragma unroll
    for (int j = 0; j < 8; ++j) {
        float4 v = src[j * 32 + lane];
        us4 o;
        o[0] = f2bf(v.x); o[1] = f2bf(v.y); o[2] = f2bf(v.z); o[3] = f2bf(v.w);
        dst[j * 32 + lane] = o;
    }
}

// ---------------------------------------------------------------------------
// 4/5) Batched expert GEMM: Out[e] = act(A[e] @ B[e] + bias[e])
//      A: [E][CAP_PAD][KDIM] bf16, B: [E][KDIM][N] f32 (converted on stage).
//      128x128 block tile, KT=32, 8 waves x (2x4) WMMA tiles, double-buffered
//      LDS; A staged with async global->LDS, B f32 loads hidden behind WMMAs.
// ---------------------------------------------------------------------------
template <int KDIM, int N, bool GELU, bool OUT_BF16>
__global__ __launch_bounds__(256) void gemm_kernel(
    const unsigned short* __restrict__ A,
    const float* __restrict__ Bsrc,
    const float* __restrict__ bias,
    void* __restrict__ Out)
{
    constexpr int LDSP  = 40;             // padded row stride (ushorts), 80B
    constexpr int NSTEP = KDIM / 32;
    __shared__ alignas(16) unsigned short As[2][128 * LDSP];
    __shared__ alignas(16) unsigned short Bs[2][128 * LDSP];

    const int e  = blockIdx.z;
    const int Mb = blockIdx.y * 128;
    const int Nb = blockIdx.x * 128;

    const unsigned short* Ae = A    + (size_t)e * CAP_PAD * KDIM;
    const float*          Be = Bsrc + (size_t)e * KDIM * N;
    const float*          be = bias + (size_t)e * N;

    const int tid   = threadIdx.x;
    const int wave  = tid >> 5;
    const int lane  = tid & 31;
    const int waveM = wave & 3;           // 4 waves along M
    const int waveN = wave >> 2;          // 2 waves along N
    const int half  = lane >> 4;
    const int l16   = lane & 15;

    // staging index split
    const int ar  = tid >> 2;             // A: rows (0..63), two passes
    const int acq = tid & 3;              //    8-ushort column quarter
    const int bk8 = tid >> 5;             // B: k sub-row (0..7), four passes
    const int bnq = tid & 31;             //    4-float column group

    v8f acc[2][4];
    const v8f zero8 = {0.f, 0.f, 0.f, 0.f, 0.f, 0.f, 0.f, 0.f};
#pragma unroll
    for (int mt = 0; mt < 2; ++mt)
#pragma unroll
        for (int nt = 0; nt < 4; ++nt) acc[mt][nt] = zero8;

    // ---- prologue: stage k-step 0 into buffer 0 ----
#pragma unroll
    for (int pp = 0; pp < 2; ++pp) {
        int row = pp * 64 + ar;
        copy16_g2l(Ae + (size_t)(Mb + row) * KDIM + acq * 8,
                   &As[0][row * LDSP + acq * 8]);
    }
    {
        float4 bv[4];
#pragma unroll
        for (int pp = 0; pp < 4; ++pp)
            bv[pp] = *(const float4*)(Be + (size_t)(pp * 8 + bk8) * N + Nb + bnq * 4);
#pragma unroll
        for (int pp = 0; pp < 4; ++pp) {
            int kk = pp * 8 + bk8, c = bnq * 4;
            Bs[0][(c + 0) * LDSP + kk] = f2bf(bv[pp].x);
            Bs[0][(c + 1) * LDSP + kk] = f2bf(bv[pp].y);
            Bs[0][(c + 2) * LDSP + kk] = f2bf(bv[pp].z);
            Bs[0][(c + 3) * LDSP + kk] = f2bf(bv[pp].w);
        }
    }
    wait_async_done();
    __syncthreads();

    // ---- main loop: compute buffer cur while staging buffer nxt ----
    for (int step = 0; step < NSTEP; ++step) {
        const int  cur      = step & 1;
        const int  nxt      = cur ^ 1;
        const bool has_next = (step + 1) < NSTEP;
        const int  k0n      = (step + 1) * 32;

        float4 bv[4];
        if (has_next) {
            // fire-and-forget async copies for next A tile
#pragma unroll
            for (int pp = 0; pp < 2; ++pp) {
                int row = pp * 64 + ar;
                copy16_g2l(Ae + (size_t)(Mb + row) * KDIM + k0n + acq * 8,
                           &As[nxt][row * LDSP + acq * 8]);
            }
            // next B f32 loads in flight; consumed after the WMMA block
#pragma unroll
            for (int pp = 0; pp < 4; ++pp)
                bv[pp] = *(const float4*)(Be + (size_t)(k0n + pp * 8 + bk8) * N + Nb + bnq * 4);
        }

        // fragments (ISA 16-bit A 16x32 layout: half selects K base 0/8,
        // runs at K and K+16; B mirrored with the same K layout)
        FragBF16 a[2], b[4];
#pragma unroll
        for (int mt = 0; mt < 2; ++mt) {
            int row = waveM * 32 + mt * 16 + l16;
            const unsigned short* p = &As[cur][row * LDSP + half * 8];
            a[mt].u[0] = *(const us8*)p;
            a[mt].u[1] = *(const us8*)(p + 16);
        }
#pragma unroll
        for (int nt = 0; nt < 4; ++nt) {
            int col = waveN * 64 + nt * 16 + l16;
            const unsigned short* p = &Bs[cur][col * LDSP + half * 8];
            b[nt].u[0] = *(const us8*)p;
            b[nt].u[1] = *(const us8*)(p + 16);
        }
#pragma unroll
        for (int mt = 0; mt < 2; ++mt)
#pragma unroll
            for (int nt = 0; nt < 4; ++nt)
                acc[mt][nt] = __builtin_amdgcn_wmma_f32_16x16x32_bf16(
                    false, a[mt].v, false, b[nt].v,
                    (short)0, acc[mt][nt], false, false);

        if (has_next) {
#pragma unroll
            for (int pp = 0; pp < 4; ++pp) {
                int kk = pp * 8 + bk8, c = bnq * 4;
                Bs[nxt][(c + 0) * LDSP + kk] = f2bf(bv[pp].x);
                Bs[nxt][(c + 1) * LDSP + kk] = f2bf(bv[pp].y);
                Bs[nxt][(c + 2) * LDSP + kk] = f2bf(bv[pp].z);
                Bs[nxt][(c + 3) * LDSP + kk] = f2bf(bv[pp].w);
            }
        }
        wait_async_done();
        __syncthreads();
    }

    // ---- epilogue: bias (+GELU) and store; C layout: m = v + half*8, n = l16
#pragma unroll
    for (int mt = 0; mt < 2; ++mt) {
#pragma unroll
        for (int nt = 0; nt < 4; ++nt) {
            int n = Nb + waveN * 64 + nt * 16 + l16;
            float bvs = be[n];
#pragma unroll
            for (int v = 0; v < 8; ++v) {
                int m = Mb + waveM * 32 + mt * 16 + half * 8 + v;
                float val = acc[mt][nt][v] + bvs;
                if (GELU) val = gelu_tanh(val);
                size_t idx = (size_t)e * CAP_PAD * N + (size_t)m * N + n;
                if (OUT_BF16) ((unsigned short*)Out)[idx] = f2bf(val);
                else          ((float*)Out)[idx] = val;
            }
        }
    }
}

// ---------------------------------------------------------------------------
// 6) Gather: out[t] = g0*ye[e0][p0] + g1*ye[e1][p1]  (dropped -> gate 0).
// ---------------------------------------------------------------------------
__global__ __launch_bounds__(256) void gather_kernel(
    const float* __restrict__ ye,
    const int* __restrict__ slot_expert, const int* __restrict__ slot_pos,
    const float* __restrict__ slot_gate, float* __restrict__ out)
{
    int wave = threadIdx.x >> 5;
    int lane = threadIdx.x & 31;
    int t = blockIdx.x * 8 + wave;

    int e0 = slot_expert[t],          p0 = slot_pos[t];
    int e1 = slot_expert[T_TOK + t],  p1 = slot_pos[T_TOK + t];
    float g0 = slot_gate[t], g1 = slot_gate[T_TOK + t];

    const float4* r0 = (const float4*)(ye + ((size_t)e0 * CAP_PAD + p0) * DIM);
    const float4* r1 = (const float4*)(ye + ((size_t)e1 * CAP_PAD + p1) * DIM);
    float4* o = (float4*)(out + (size_t)t * DIM);
#pragma unroll
    for (int j = 0; j < 8; ++j) {
        float4 a = r0[j * 32 + lane];
        float4 b = r1[j * 32 + lane];
        float4 r;
        r.x = g0 * a.x + g1 * b.x;
        r.y = g0 * a.y + g1 * b.y;
        r.z = g0 * a.z + g1 * b.z;
        r.w = g0 * a.w + g1 * b.w;
        o[j * 32 + lane] = r;
    }
}

// ---------------------------------------------------------------------------
extern "C" void kernel_launch(void* const* d_in, const int* in_sizes, int n_in,
                              void* d_out, int out_size, void* d_ws, size_t ws_size,
                              hipStream_t stream)
{
    const float* x  = (const float*)d_in[0];
    const float* Wg = (const float*)d_in[1];
    const float* bg = (const float*)d_in[2];
    const float* W1 = (const float*)d_in[3];
    const float* b1 = (const float*)d_in[4];
    const float* W2 = (const float*)d_in[5];
    const float* b2 = (const float*)d_in[6];
    float* out = (float*)d_out;

    char* ws = (char*)d_ws;
    int*   slot_expert = (int*)(ws);
    int*   slot_pos    = (int*)(ws + 64 * 1024);
    float* slot_gate   = (float*)(ws + 128 * 1024);
    unsigned short* buf = (unsigned short*)(ws + 256 * 1024);
    size_t buf_bytes = (size_t)EXPERTS * CAP_PAD * DIM * 2;     // 20 MB
    unsigned short* h  = (unsigned short*)((char*)buf + buf_bytes);
    size_t h_bytes = (size_t)EXPERTS * CAP_PAD * FDIM * 2;      // 80 MB
    float* ye = (float*)((char*)h + h_bytes);                   // 40 MB

    gating_kernel<<<T_TOK / 8, 256, 0, stream>>>(x, Wg, bg, slot_expert, slot_gate);
    scan_kernel<<<EXPERTS, 1024, 0, stream>>>(slot_expert, slot_pos, slot_gate);
    zero_kernel<<<2560, 256, 0, stream>>>((uint4*)buf, (long)(buf_bytes / 16));
    dispatch_kernel<<<(T_TOK * TOPK) / 8, 256, 0, stream>>>(x, slot_expert, slot_pos, buf);

    dim3 g1(FDIM / 128, CAP_PAD / 128, EXPERTS);                // 32 x 10 x 8
    gemm_kernel<DIM, FDIM, true, true><<<g1, 256, 0, stream>>>(buf, W1, b1, (void*)h);

    dim3 g2(DIM / 128, CAP_PAD / 128, EXPERTS);                 // 8 x 10 x 8
    gemm_kernel<FDIM, DIM, false, false><<<g2, 256, 0, stream>>>(h, W2, b2, (void*)ye);

    gather_kernel<<<T_TOK / 8, 256, 0, stream>>>(ye, slot_expert, slot_pos, slot_gate, out);
}
